// DetNet_68410239091161
// MI455X (gfx1250) — matrix operand order repaired and checked by reading, not compiled, not run on hardware
//
#include <hip/hip_runtime.h>
#include <stdint.h>

#define LAYERS 10
#define SYM    128
#define OHN    16
#define X2     256      // 2*SYM
#define MDIM   256      // field samples
#define IN1    768      // 2*(SYM+V_LEN)
#define HID    1024     // 2*Z_LEN
#define OUT2   2048     // SYM*OH
#define OUT3   512      // 2*V_LEN
#define BATCH  1024

typedef __attribute__((ext_vector_type(16))) __bf16 v16bf;
typedef __attribute__((ext_vector_type(8)))  float  v8f;
typedef __attribute__((ext_vector_type(4)))  int    v4i_t;

// f32x2 -> packed bf16x2 (RNE): two rounding adds + one v_perm_b32
__device__ __forceinline__ uint32_t pack_bf16x2(float lo, float hi) {
    uint32_t a = __float_as_uint(lo), b = __float_as_uint(hi);
    a += 0x7FFFu + ((a >> 16) & 1u);
    b += 0x7FFFu + ((b >> 16) & 1u);
    return __builtin_amdgcn_perm(b, a, 0x07060302);   // {b[31:16], a[31:16]}
}

// ---------------- async global->LDS copy (CDNA5 ASYNCcnt path), with fallback
#if defined(__has_builtin)
#if __has_builtin(__builtin_amdgcn_global_load_async_to_lds_b128) && \
    __has_builtin(__builtin_amdgcn_s_wait_asynccnt)
#define ASYNC_LDS 1
#endif
#endif

typedef __attribute__((address_space(1))) void*  gvoidp;
typedef __attribute__((address_space(3))) void*  lvoidp;
typedef __attribute__((address_space(1))) v4i_t* gv4p;
typedef __attribute__((address_space(3))) v4i_t* lv4p;

__device__ __forceinline__ void cp_b128(const void* g, void* lds) {
#ifdef ASYNC_LDS
    __builtin_amdgcn_global_load_async_to_lds_b128(
        (gv4p)(gvoidp)(void*)g, (lv4p)(lvoidp)lds, 0, 0);
#else
    *(uint4*)lds = *(const uint4*)g;
#endif
}
__device__ __forceinline__ void cp_wait() {
#ifdef ASYNC_LDS
    __builtin_amdgcn_s_wait_asynccnt(0);
#endif
}

// ---------------------------------------------------------------- zero init
__global__ void zero_kernel(float* p, int n) {
    int i = blockIdx.x * 256 + threadIdx.x;
    if (i < n) p[i] = 0.f;
}

// ---------------------------------------------------------------- field part
// One block per (batch b, which in {e,o}). Computes
//   contrib = Psi^T @ ( -c1 * (Psi@x) * y + c2 * (Psi@x)^3 )
// streaming Psi from HBM in 32-row LDS-staged chunks via async-to-LDS.
__global__ __launch_bounds__(256)
void field_kernel(const float* __restrict__ Psi_e, const float* __restrict__ Psi_o,
                  const float* __restrict__ y_e,   const float* __restrict__ y_o,
                  const float* __restrict__ x,     const float* __restrict__ deltas,
                  float* __restrict__ qpart, int layer)
{
    __shared__ __align__(16) float psi_s[32 * 260];   // 32 rows, padded stride
    __shared__ float x_s[256];
    __shared__ float s_s[256];
    __shared__ float u_s[256];

    const int b = blockIdx.x;
    const int which = blockIdx.y;
    const int t = threadIdx.x;

    const float* Psi = (which ? Psi_o : Psi_e) + (size_t)b * MDIM * X2;
    const float* y   = (which ? y_o   : y_e)   + (size_t)b * MDIM;
    const float c1 = which ? deltas[layer*4 + 1] : deltas[layer*4 + 0];
    const float c2 = which ? deltas[layer*4 + 3] : deltas[layer*4 + 1];

    x_s[t] = x[(size_t)b * X2 + t];
    __syncthreads();

    // pass 1: s[m] = dot(Psi[m,:], x)
    for (int ch = 0; ch < 8; ++ch) {
        const float* base = Psi + (size_t)ch * 32 * X2;
        if (ch < 7) __builtin_prefetch(Psi + (size_t)(ch + 1) * 32 * X2 + t * 16, 0, 1);
        #pragma unroll
        for (int i = 0; i < 8; ++i) {
            int f = t + i * 256;                 // 2048 float4 tiles
            int row = f >> 6, q = f & 63;
            cp_b128(base + row * 256 + q * 4, &psi_s[row * 260 + q * 4]);
        }
        cp_wait();
        __syncthreads();
        int r = t >> 3, seg = t & 7;
        const float* prow = psi_s + r * 260 + seg * 32;
        const float* xp   = x_s + seg * 32;
        float p = 0.f;
        #pragma unroll
        for (int j = 0; j < 32; ++j) p += prow[j] * xp[j];
        p += __shfl_xor(p, 1);
        p += __shfl_xor(p, 2);
        p += __shfl_xor(p, 4);
        if (seg == 0) s_s[ch * 32 + r] = p;
        __syncthreads();
    }

    { float s = s_s[t]; u_s[t] = -c1 * s * y[t] + c2 * s * s * s; }
    __syncthreads();

    // pass 2: contrib[l] = sum_m Psi[m,l] * u[m]
    float acc = 0.f;
    for (int ch = 0; ch < 8; ++ch) {
        const float* base = Psi + (size_t)ch * 32 * X2;
        #pragma unroll
        for (int i = 0; i < 8; ++i) {
            int f = t + i * 256;
            int row = f >> 6, q = f & 63;
            cp_b128(base + row * 256 + q * 4, &psi_s[row * 260 + q * 4]);
        }
        cp_wait();
        __syncthreads();
        const float* up = u_s + ch * 32;
        #pragma unroll
        for (int m = 0; m < 32; ++m) acc += psi_s[m * 260 + t] * up[m];
        __syncthreads();
    }
    qpart[((size_t)which * BATCH + b) * X2 + t] = acc;
}

// ---------------------------------------------------------------- pack [q|v] -> bf16
__global__ __launch_bounds__(256)
void pack_kernel(const float* __restrict__ x, const float* __restrict__ qpart,
                 const float* __restrict__ v, uint32_t* __restrict__ in1)
{
    int idx = blockIdx.x * 256 + threadIdx.x;     // over BATCH*IN1/2 bf16 pairs
    int b = idx / (IN1 / 2), p = idx % (IN1 / 2);
    int c = p * 2;
    float v0, v1;
    if (c < X2) {
        v0 = x[(size_t)b * X2 + c]     + qpart[(size_t)b * X2 + c]
           + qpart[(size_t)(BATCH + b) * X2 + c];
        v1 = x[(size_t)b * X2 + c + 1] + qpart[(size_t)b * X2 + c + 1]
           + qpart[(size_t)(BATCH + b) * X2 + c + 1];
    } else {
        v0 = v[(size_t)b * OUT3 + (c - X2)];
        v1 = v[(size_t)b * OUT3 + (c - X2) + 1];
    }
    in1[idx] = pack_bf16x2(v0, v1);
}

// ---------------------------------------------------------------- WMMA GEMM
// C(BATCH x N) = A(BATCH x K, bf16) @ W(N x K, f32->bf16)^T  + epilogue
// MODE 0: z = relu(C+b) -> bf16 ; MODE 1: xoh += C+b, softmax(16) -> x ; MODE 2: v += C+b
template<int MODE, int N, int K>
__global__ __launch_bounds__(256)
void gemm_kernel(const uint16_t* __restrict__ A, const float* __restrict__ W,
                 const float* __restrict__ bias,
                 uint32_t* __restrict__ zout, float* __restrict__ xoh,
                 float* __restrict__ vws, float* __restrict__ xws,
                 float* __restrict__ out_xs, float* __restrict__ out_xoh,
                 const float* __restrict__ mapp_re, const float* __restrict__ mapp_im)
{
    __shared__ __align__(16) uint32_t a_s[64 * 20];  // 64 rows x 32 bf16 (padded)
    __shared__ __align__(16) uint32_t w_s[64 * 20];
    __shared__ float c_s[64 * 65];

    const int t = threadIdx.x;
    const int wave = t >> 5, lane = t & 31;
    const int bm0 = blockIdx.x * 64;
    const int bn0 = blockIdx.y * 64;
    const int wm = (wave >> 1) * 16;   // 0/16/32/48
    const int wn = (wave & 1) * 32;    // 0/32

    v8f acc0 = {}, acc1 = {};

    for (int kt = 0; kt < K; kt += 32) {
        {   // stage A: one b128 (8 bf16) per thread, async -> LDS
            int row = t >> 2, q = t & 3;
            cp_b128(A + (size_t)(bm0 + row) * K + kt + q * 8, &a_s[row * 20 + q * 4]);
        }
        #pragma unroll
        for (int i = 0; i < 2; ++i) {   // stage W: f32 -> bf16 via VGPRs
            int f = t + i * 256;
            int row = f >> 3, q = f & 7;
            float4 wv = *(const float4*)(W + (size_t)(bn0 + row) * K + kt + q * 4);
            w_s[row * 20 + q * 2]     = pack_bf16x2(wv.x, wv.y);
            w_s[row * 20 + q * 2 + 1] = pack_bf16x2(wv.z, wv.w);
        }
        cp_wait();
        __syncthreads();

        union Frag { uint32_t u[8]; v16bf v; } fa, fb0, fb1;
        {   // A frag: lanes<16 -> K dwords 0..7, lanes>=16 -> 8..15
            int base = (wm + (lane & 15)) * 20 + ((lane < 16) ? 0 : 8);
            #pragma unroll
            for (int j = 0; j < 8; ++j) { fa.u[j] = a_s[base + j]; }
        }
        {   // B frags: lanes<16 -> K dwords 0..7, lanes>=16 -> 8..15
            int base0 = (wn +      (lane & 15)) * 20 + ((lane < 16) ? 0 : 8);
            int base1 = (wn + 16 + (lane & 15)) * 20 + ((lane < 16) ? 0 : 8);
            #pragma unroll
            for (int j = 0; j < 8; ++j) { fb0.u[j] = w_s[base0 + j]; fb1.u[j] = w_s[base1 + j]; }
        }
        acc0 = __builtin_amdgcn_wmma_f32_16x16x32_bf16(false, fa.v, false, fb0.v,
                                                       (short)0, acc0, false, false);
        acc1 = __builtin_amdgcn_wmma_f32_16x16x32_bf16(false, fa.v, false, fb1.v,
                                                       (short)0, acc1, false, false);
        __syncthreads();
    }

    {   // accumulators -> c_s (C layout: VGPR v = row v / v+8, lane = col)
        int col = wn + (lane & 15);
        int rbase = wm + ((lane < 16) ? 0 : 8);
        #pragma unroll
        for (int vv = 0; vv < 8; ++vv) {
            c_s[(rbase + vv) * 65 + col]      = acc0[vv];
            c_s[(rbase + vv) * 65 + col + 16] = acc1[vv];
        }
    }
    __syncthreads();

    if (MODE == 0) {
        #pragma unroll
        for (int i = 0; i < 8; ++i) {       // paired bf16 stores (u32)
            int idx = t + i * 256;          // 2048 pairs in 64x64 tile
            int r = idx >> 5, cp2 = idx & 31;
            int c = cp2 * 2;
            int grow = bm0 + r, gcol = bn0 + c;
            float v0 = c_s[r * 65 + c]     + bias[gcol];
            float v1 = c_s[r * 65 + c + 1] + bias[gcol + 1];
            v0 = v0 > 0.f ? v0 : 0.f;
            v1 = v1 > 0.f ? v1 : 0.f;
            zout[((size_t)grow * N + gcol) >> 1] = pack_bf16x2(v0, v1);
        }
    } else if (MODE == 2) {
        #pragma unroll
        for (int i = 0; i < 16; ++i) {
            int idx = t + i * 256;
            int r = idx >> 6, c = idx & 63;
            int grow = bm0 + r, gcol = bn0 + c;
            float val = c_s[r * 65 + c] + bias[gcol] + vws[(size_t)grow * N + gcol];
            vws[(size_t)grow * N + gcol] = val;
        }
    } else {
        #pragma unroll
        for (int i = 0; i < 16; ++i) {   // finalize running x_oh
            int idx = t + i * 256;
            int r = idx >> 6, c = idx & 63;
            int grow = bm0 + r, gcol = bn0 + c;
            float val = c_s[r * 65 + c] + bias[gcol] + xoh[(size_t)grow * N + gcol];
            c_s[r * 65 + c] = val;
            xoh[(size_t)grow * N + gcol]     = val;
            out_xoh[(size_t)grow * N + gcol] = val;
        }
        __syncthreads();
        // softmax over 16-wide groups + constellation dot -> x
        int r = t >> 2, g = t & 3;                // 64 rows x 4 groups
        const float* p = &c_s[r * 65 + g * 16];
        float mx = p[0];
        #pragma unroll
        for (int j = 1; j < 16; ++j) mx = fmaxf(mx, p[j]);
        float se = 0.f, re = 0.f, im = 0.f;
        #pragma unroll
        for (int j = 0; j < 16; ++j) {
            float e = __expf(p[j] - mx);
            se += e; re += e * mapp_re[j]; im += e * mapp_im[j];
        }
        re /= se; im /= se;
        int b = bm0 + r;
        int G = (bn0 >> 4) + g;
        xws[(size_t)b * X2 + G]          = re;
        xws[(size_t)b * X2 + 128 + G]    = im;
        out_xs[(size_t)b * X2 + G]       = re;
        out_xs[(size_t)b * X2 + 128 + G] = im;
    }
}

// ---------------------------------------------------------------- driver
extern "C" void kernel_launch(void* const* d_in, const int* in_sizes, int n_in,
                              void* d_out, int out_size, void* d_ws, size_t ws_size,
                              hipStream_t stream)
{
    const float* y_e     = (const float*)d_in[0];
    const float* y_o     = (const float*)d_in[1];
    const float* Psi_e   = (const float*)d_in[2];
    const float* Psi_o   = (const float*)d_in[3];
    const float* mapp_re = (const float*)d_in[4];
    const float* mapp_im = (const float*)d_in[5];
    const float* W1      = (const float*)d_in[6];
    const float* b1      = (const float*)d_in[7];
    const float* W2      = (const float*)d_in[8];
    const float* b2      = (const float*)d_in[9];
    const float* W3      = (const float*)d_in[10];
    const float* b3      = (const float*)d_in[11];
    const float* deltas  = (const float*)d_in[12];
    (void)in_sizes; (void)n_in; (void)out_size; (void)ws_size;

    float* x_ws   = (float*)d_ws;                       // B*256  (starts the zeroed span)
    float* v_ws   = x_ws + (size_t)BATCH * X2;          // B*512
    float* xoh_ws = v_ws + (size_t)BATCH * OUT3;        // B*2048
    float* qpart  = xoh_ws + (size_t)BATCH * OUT2;      // 2*B*256
    uint32_t* in1 = (uint32_t*)(qpart + 2 * (size_t)BATCH * X2);    // B*768 bf16
    uint16_t* z   = (uint16_t*)(in1 + (size_t)BATCH * IN1 / 2);     // B*1024 bf16

    float* out_xs  = (float*)d_out;                          // (10,B,256)
    float* out_xoh = out_xs + (size_t)LAYERS * BATCH * X2;   // (10,B,2048)

    const int nz = BATCH * (X2 + OUT3 + OUT2);               // x|v|xoh contiguous
    zero_kernel<<<(nz + 255) / 256, 256, 0, stream>>>(x_ws, nz);

    for (int l = 0; l < LAYERS; ++l) {
        field_kernel<<<dim3(BATCH, 2), 256, 0, stream>>>(
            Psi_e, Psi_o, y_e, y_o, x_ws, deltas, qpart, l);

        pack_kernel<<<(BATCH * IN1 / 2) / 256, 256, 0, stream>>>(x_ws, qpart, v_ws, in1);

        gemm_kernel<0, HID, IN1><<<dim3(BATCH / 64, HID / 64), 256, 0, stream>>>(
            (const uint16_t*)in1, W1 + (size_t)l * HID * IN1, b1 + (size_t)l * HID,
            (uint32_t*)z, nullptr, nullptr, nullptr, nullptr, nullptr, mapp_re, mapp_im);

        gemm_kernel<1, OUT2, HID><<<dim3(BATCH / 64, OUT2 / 64), 256, 0, stream>>>(
            z, W2 + (size_t)l * OUT2 * HID, b2 + (size_t)l * OUT2,
            nullptr, xoh_ws, nullptr, x_ws,
            out_xs + (size_t)l * BATCH * X2,
            out_xoh + (size_t)l * BATCH * OUT2, mapp_re, mapp_im);

        gemm_kernel<2, OUT3, HID><<<dim3(BATCH / 64, OUT3 / 64), 256, 0, stream>>>(
            z, W3 + (size_t)l * OUT3 * HID, b3 + (size_t)l * OUT3,
            nullptr, nullptr, v_ws, nullptr, nullptr, nullptr, mapp_re, mapp_im);
    }
}
